// GCN_17145509445674
// MI455X (gfx1250) — matrix operand order repaired and checked by reading, not compiled, not run on hardware
//
#include <hip/hip_runtime.h>

typedef __attribute__((ext_vector_type(16))) _Float16 v16h;
typedef __attribute__((ext_vector_type(8)))  _Float16 v8h;
typedef __attribute__((ext_vector_type(8)))  float    v8f;

// ---------------------------------------------------------------------------
// Degree / normalization: dis[i] = rsqrt(deg[i]), deg includes self-loop (>=1)
// ---------------------------------------------------------------------------
__global__ void deg_init_kernel(float* __restrict__ deg, int n) {
    int i = blockIdx.x * blockDim.x + threadIdx.x;
    if (i < n) deg[i] = 1.0f;   // self-loop contribution
}

__global__ void deg_count_kernel(const int* __restrict__ dst,
                                 float* __restrict__ deg, int nE) {
    int e = blockIdx.x * blockDim.x + threadIdx.x;
    if (e < nE) atomicAdd(&deg[dst[e]], 1.0f);
}

__global__ void deg_rsqrt_kernel(float* __restrict__ deg, int n) {
    int i = blockIdx.x * blockDim.x + threadIdx.x;
    if (i < n) deg[i] = rsqrtf(deg[i]);   // deg >= 1 always
}

// ---------------------------------------------------------------------------
// WMMA GEMM: OUT[M x 64] = X[M x K] * W[K x 64], fp32 in/out, f16 WMMA math.
// K is a compile-time constant (128 or 64) so the k-loop fully unrolls.
//
// - W is converted to f16 once per block into LDS, TRANSPOSED: WT[n][k],
//   padded by 8 halfs/row (16 B) so ds_load_b128 stays aligned and the 16
//   columns of a half-wave land on distinct banks (row stride = 68 dwords).
// - One wave computes a 16-row stripe across ALL four 16-col tiles: each A
//   fragment is reused by 4 consecutive v_wmma ops.
// Requires M % 16 == 0 (M = 100000).
// ---------------------------------------------------------------------------
template <int K>
__global__ void gemm_wmma_kernel(const float* __restrict__ X,
                                 const float* __restrict__ W,
                                 float* __restrict__ OUT, int M) {
    __shared__ _Float16 WT[64][K + 8];   // transposed, padded f16 copy of W

    // Cooperative fill: coalesced read of W[k][n], scatter to WT[n][k].
    for (int idx = threadIdx.x; idx < 64 * K; idx += blockDim.x) {
        int k = idx >> 6;            // idx = k*64 + n
        int n = idx & 63;
        WT[n][k] = (_Float16)W[idx];
    }
    __syncthreads();

    const int lane  = threadIdx.x & 31;
    const int wave  = blockIdx.x * (blockDim.x >> 5) + (threadIdx.x >> 5);
    const int mtile = wave;                       // one 16-row stripe per wave
    if (mtile * 16 >= M) return;                  // wave-uniform guard

    const int hi    = lane >> 4;                  // 0 or 1
    const int l15   = lane & 15;
    const int arow  = mtile * 16 + l15;           // A-fragment row
    const int khalf = hi * 8;                     // A-frag K sub-offset
    const int kbB   = hi * 16;                    // unused: B frag uses khalf, like A

    v8f acc[4] = {};

    #pragma unroll
    for (int k0 = 0; k0 < K; k0 += 32) {
        // ---- A fragment: K = k0+khalf+{0..7} and k0+khalf+16+{0..7}
        const float* xa = X + (long)arow * K + k0 + khalf;
        const float4 a0 = *(const float4*)(xa + 0);
        const float4 a1 = *(const float4*)(xa + 4);
        const float4 a2 = *(const float4*)(xa + 16);
        const float4 a3 = *(const float4*)(xa + 20);
        v16h a;
        a[0]  = (_Float16)a0.x; a[1]  = (_Float16)a0.y;
        a[2]  = (_Float16)a0.z; a[3]  = (_Float16)a0.w;
        a[4]  = (_Float16)a1.x; a[5]  = (_Float16)a1.y;
        a[6]  = (_Float16)a1.z; a[7]  = (_Float16)a1.w;
        a[8]  = (_Float16)a2.x; a[9]  = (_Float16)a2.y;
        a[10] = (_Float16)a2.z; a[11] = (_Float16)a2.w;
        a[12] = (_Float16)a3.x; a[13] = (_Float16)a3.y;
        a[14] = (_Float16)a3.z; a[15] = (_Float16)a3.w;

        // ---- 4 N-tiles share this A fragment
        #pragma unroll
        for (int nt = 0; nt < 4; ++nt) {
            const _Float16* wrow = &WT[nt * 16 + l15][k0 + khalf];
            v8h blo = *(const v8h*)(wrow);        // 16-B aligned ds loads
            v8h bhi = *(const v8h*)(wrow + 16);
            v16h b;
            #pragma unroll
            for (int e = 0; e < 8; ++e) { b[e] = blo[e]; b[8 + e] = bhi[e]; }

            acc[nt] = __builtin_amdgcn_wmma_f32_16x16x32_f16(
                          /*neg_a=*/false, a, /*neg_b=*/false, b,
                          /*c_mod=*/(short)0, acc[nt],
                          /*reuse_a=*/false, /*reuse_b=*/false);
        }
    }

    // ---- Store: element r -> row (mtile*16 + hi*8 + r), col nt*16 + l15
    const int rbase = mtile * 16 + hi * 8;
    #pragma unroll
    for (int nt = 0; nt < 4; ++nt) {
        #pragma unroll
        for (int r = 0; r < 8; ++r)
            OUT[(long)(rbase + r) * 64 + nt * 16 + l15] = acc[nt][r];
    }
}

// ---------------------------------------------------------------------------
// out[i,:] = h[i,:] * dis[i]^2   (self-loop term; non-atomic full init of out)
// ---------------------------------------------------------------------------
__global__ void selfloop_init_kernel(const float* __restrict__ h,
                                     const float* __restrict__ dis,
                                     float* __restrict__ out, int n) {
    int t = blockIdx.x * blockDim.x + threadIdx.x;
    if (t >= n * 16) return;
    int node = t >> 4;
    float d = dis[node];
    float s = d * d;
    float4 v = ((const float4*)h)[t];
    float4 r; r.x = v.x * s; r.y = v.y * s; r.z = v.z * s; r.w = v.w * s;
    ((float4*)out)[t] = r;
}

// ---------------------------------------------------------------------------
// Edge scatter: out[dst,:] += h[src,:] * dis[src]*dis[dst]
// 16 threads per edge, one float4 gather + 4 clustered f32 atomics each.
// ---------------------------------------------------------------------------
__global__ void edge_scatter_kernel(const float* __restrict__ h,
                                    const float* __restrict__ dis,
                                    const int* __restrict__ src,
                                    const int* __restrict__ dst,
                                    float* __restrict__ out, int nE) {
    long t = (long)blockIdx.x * blockDim.x + threadIdx.x;
    if (t >= (long)nE * 16) return;
    int e  = (int)(t >> 4);
    int fc = ((int)t & 15) * 4;
    int s  = src[e];
    int d  = dst[e];
    float norm = dis[s] * dis[d];
    float4 v = *(const float4*)(h + (long)s * 64 + fc);
    float* o = out + (long)d * 64 + fc;
    atomicAdd(o + 0, v.x * norm);
    atomicAdd(o + 1, v.y * norm);
    atomicAdd(o + 2, v.z * norm);
    atomicAdd(o + 3, v.w * norm);
}

// ---------------------------------------------------------------------------
// out = relu(out + bias)  over n*64 elements
// ---------------------------------------------------------------------------
__global__ void bias_relu_kernel(float* __restrict__ out,
                                 const float* __restrict__ bias, int n) {
    int i = blockIdx.x * blockDim.x + threadIdx.x;
    if (i >= n * 64) return;
    float v = out[i] + bias[i & 63];
    out[i] = v > 0.0f ? v : 0.0f;
}

// ---------------------------------------------------------------------------
// Final linear: out[i] = dot(h[i,:64], W3) + b3
// ---------------------------------------------------------------------------
__global__ void final_linear_kernel(const float* __restrict__ h,
                                    const float* __restrict__ W3,
                                    const float* __restrict__ b3,
                                    float* __restrict__ out, int n) {
    int i = blockIdx.x * blockDim.x + threadIdx.x;
    if (i >= n) return;
    const float4* hv = (const float4*)(h + (long)i * 64);
    const float4* wv = (const float4*)W3;
    float acc = 0.0f;
    #pragma unroll
    for (int k = 0; k < 16; ++k) {
        float4 a = hv[k];
        float4 w = wv[k];
        acc += a.x * w.x + a.y * w.y + a.z * w.z + a.w * w.w;
    }
    out[i] = acc + b3[0];
}

// ---------------------------------------------------------------------------
extern "C" void kernel_launch(void* const* d_in, const int* in_sizes, int n_in,
                              void* d_out, int out_size, void* d_ws, size_t ws_size,
                              hipStream_t stream) {
    const float* x  = (const float*)d_in[0];   // [N,128]
    const int* edge = (const int*)  d_in[1];   // [2,E] flat: src=edge, dst=edge+E
    const float* W1 = (const float*)d_in[2];   // [128,64]
    const float* b1 = (const float*)d_in[3];   // [64]
    const float* W2 = (const float*)d_in[4];   // [64,64]
    const float* b2 = (const float*)d_in[5];   // [64]
    const float* W3 = (const float*)d_in[6];   // [64,1]
    const float* b3 = (const float*)d_in[7];   // [1]

    const int N = in_sizes[0] / 128;           // 100000
    const int E = in_sizes[1] / 2;             // 1200000
    const int* src = edge;
    const int* dst = edge + E;
    float* out = (float*)d_out;

    // Workspace layout: bufA [N*64] | bufB [N*64] | dis [N]
    float* bufA = (float*)d_ws;
    float* bufB = bufA + (size_t)N * 64;
    float* dis  = bufB + (size_t)N * 64;

    const int TB = 256;
    const int waves_per_block = TB / 32;

    // --- degree / rsqrt normalization ---
    deg_init_kernel <<<(N + TB - 1) / TB, TB, 0, stream>>>(dis, N);
    deg_count_kernel<<<(E + TB - 1) / TB, TB, 0, stream>>>(dst, dis, E);
    deg_rsqrt_kernel<<<(N + TB - 1) / TB, TB, 0, stream>>>(dis, N);

    // GEMM grid: one wave per 16-row stripe (each wave covers all 64 cols)
    const int mtiles = (N + 15) / 16;          // 6250
    const int gemm_blocks = (mtiles + waves_per_block - 1) / waves_per_block;

    const int ew_blocks = (E * 16 + TB - 1) / TB;
    const int sl_blocks = (N * 16 + TB - 1) / TB;
    const int br_blocks = (N * 64 + TB - 1) / TB;

    // ---- Layer 1: h1 = x @ W1 (WMMA) ; aggregate ; +b1 ; relu ----
    gemm_wmma_kernel<128><<<gemm_blocks, TB, 0, stream>>>(x, W1, bufA, N);
    selfloop_init_kernel<<<sl_blocks, TB, 0, stream>>>(bufA, dis, bufB, N);
    edge_scatter_kernel<<<ew_blocks, TB, 0, stream>>>(bufA, dis, src, dst, bufB, E);
    bias_relu_kernel<<<br_blocks, TB, 0, stream>>>(bufB, b1, N);

    // ---- Layer 2: h2 = h1' @ W2 (WMMA) ; aggregate ; +b2 ; relu ----
    gemm_wmma_kernel<64><<<gemm_blocks, TB, 0, stream>>>(bufB, W2, bufA, N);
    selfloop_init_kernel<<<sl_blocks, TB, 0, stream>>>(bufA, dis, bufB, N);
    edge_scatter_kernel<<<ew_blocks, TB, 0, stream>>>(bufA, dis, src, dst, bufB, E);
    bias_relu_kernel<<<br_blocks, TB, 0, stream>>>(bufB, b2, N);

    // ---- Layer 3: out = h2' @ W3 + b3 ----
    final_linear_kernel<<<(N + TB - 1) / TB, TB, 0, stream>>>(bufB, W3, b3, out, N);
}
